// EfficientMOELayer_7705171329368
// MI455X (gfx1250) — compile-verified, matrix operand order checked by reading, not run on hardware
//
#include <hip/hip_runtime.h>
#include <hip/hip_bf16.h>

// ---------------------------------------------------------------------------
// MoE layer for MI455X (gfx1250): bf16 WMMA GEMMs with fp32 accumulation,
// LDS-staged tiles (128x256 block, 64x64 wave tile, 16 WMMAs per k-step).
// ---------------------------------------------------------------------------

typedef __attribute__((ext_vector_type(16))) __bf16 v16bf;
typedef __attribute__((ext_vector_type(8)))  __bf16 v8bf;
typedef __attribute__((ext_vector_type(8)))  float  v8f;

#define E_EXPERTS 16
#define TOPK      2
#define CAP       1280
#define DIM       1024
#define FFE       512    // expert FF dim
#define DFFS      512    // shared FF dim
#define NTOK      8192   // B*T

#define LDS_STRIDE 40    // bf16 elements per LDS tile row (80B: 16B-aligned, conflict-free)

// ---------------------------------------------------------------- utilities
__global__ void fill_u32_kernel(unsigned* __restrict__ p, unsigned val, size_t n) {
    size_t i = (size_t)blockIdx.x * blockDim.x + threadIdx.x;
    size_t stride = (size_t)gridDim.x * blockDim.x;
    for (; i < n; i += stride) p[i] = val;
}

__global__ void f32_to_bf16_kernel(const float* __restrict__ src, __bf16* __restrict__ dst, size_t n) {
    size_t i = (size_t)blockIdx.x * blockDim.x + threadIdx.x;
    size_t stride = (size_t)gridDim.x * blockDim.x;
    for (; i < n; i += stride) dst[i] = (__bf16)src[i];
}

// Transpose (R x Ccols) fp32 -> (Ccols x R) bf16, batched over blockIdx.z.
__global__ __launch_bounds__(256) void transpose_to_bf16_kernel(
    const float* __restrict__ src, __bf16* __restrict__ dst,
    int R, int Ccols, size_t sStride, size_t dStride)
{
    __shared__ float tile[32][33];
    src += (size_t)blockIdx.z * sStride;
    dst += (size_t)blockIdx.z * dStride;
    int c0 = blockIdx.x * 32, r0 = blockIdx.y * 32;
    int tx = threadIdx.x, ty = threadIdx.y;   // blockDim = (32, 8)
    #pragma unroll
    for (int i = ty; i < 32; i += 8)
        tile[i][tx] = src[(size_t)(r0 + i) * Ccols + (c0 + tx)];
    __syncthreads();
    #pragma unroll
    for (int i = ty; i < 32; i += 8)
        dst[(size_t)(c0 + i) * R + (r0 + tx)] = (__bf16)tile[tx][i];
}

// ---------------------------------------------------------------- router
__global__ __launch_bounds__(256) void router_kernel(
    const float* __restrict__ x, const float* __restrict__ Wr, const float* __restrict__ Wg,
    int* __restrict__ route_e, float* __restrict__ route_gate,
    float* __restrict__ gshared, float* __restrict__ imp, int* __restrict__ counts)
{
    int token = blockIdx.x * 8 + (threadIdx.x >> 5);
    int lane  = threadIdx.x & 31;
    const float* xr = x + (size_t)token * DIM;

    float acc[17];
    #pragma unroll
    for (int e = 0; e < 17; e++) acc[e] = 0.f;

    for (int d = lane; d < DIM; d += 32) {
        float xv = xr[d];
        const float* wrow = Wr + (size_t)d * E_EXPERTS;
        #pragma unroll
        for (int e = 0; e < E_EXPERTS; e++) acc[e] += xv * wrow[e];
        acc[16] += xv * Wg[d];
    }
    #pragma unroll
    for (int off = 16; off > 0; off >>= 1) {
        #pragma unroll
        for (int e = 0; e < 17; e++) acc[e] += __shfl_xor(acc[e], off, 32);
    }
    if (lane == 0) {
        float mx = acc[0];
        #pragma unroll
        for (int e = 1; e < E_EXPERTS; e++) mx = fmaxf(mx, acc[e]);
        float p[16], s = 0.f;
        #pragma unroll
        for (int e = 0; e < E_EXPERTS; e++) { p[e] = __expf(acc[e] - mx); s += p[e]; }
        float inv = 1.f / s;
        int b1 = 0; float v1 = -1.f;
        #pragma unroll
        for (int e = 0; e < E_EXPERTS; e++) { p[e] *= inv; if (p[e] > v1) { v1 = p[e]; b1 = e; } }
        int b2 = 0; float v2 = -2.f;
        #pragma unroll
        for (int e = 0; e < E_EXPERTS; e++) { if (e != b1 && p[e] > v2) { v2 = p[e]; b2 = e; } }
        #pragma unroll
        for (int e = 0; e < E_EXPERTS; e++) atomicAdd(&imp[e], p[e]);
        atomicAdd(&counts[b1], 1);
        atomicAdd(&counts[b2], 1);
        route_e[token * 2 + 0] = b1;  route_gate[token * 2 + 0] = v1;
        route_e[token * 2 + 1] = b2;  route_gate[token * 2 + 1] = v2;
        gshared[token] = 1.f / (1.f + __expf(-acc[16]));
    }
}

// ------------------------------------------------- deterministic positions
__global__ void position_kernel(
    const int* __restrict__ route_e, const float* __restrict__ route_gate,
    int* __restrict__ route_pos, int* __restrict__ tok_slot, float* __restrict__ gate_slot)
{
    int e = blockIdx.x;
    int lane = threadIdx.x;              // blockDim = 32
    int base = 0;
    for (int r0 = 0; r0 < NTOK * TOPK; r0 += 32) {
        int r = r0 + lane;
        bool m = (route_e[r] == e);
        unsigned mask = (unsigned)__ballot(m);
        if (m) {
            int pos = base + __popc(mask & ((1u << lane) - 1u));
            if (pos < CAP) {
                route_pos[r] = pos;
                tok_slot[e * CAP + pos]  = r >> 1;        // token id
                gate_slot[e * CAP + pos] = route_gate[r];
            } else {
                route_pos[r] = -1;
            }
        }
        base += __popc(mask);
    }
}

// ---------------------------------------------------------------- gather
__global__ __launch_bounds__(128) void gather_kernel(
    const __bf16* __restrict__ xbf, const int* __restrict__ route_e,
    const int* __restrict__ route_pos, __bf16* __restrict__ buf)
{
    int rt  = blockIdx.x;
    int pos = route_pos[rt];
    if (pos < 0) return;
    int e   = route_e[rt];
    int tok = rt >> 1;
    const uint4* src = (const uint4*)(xbf + (size_t)tok * DIM);
    uint4*       dst = (uint4*)(buf + ((size_t)e * CAP + pos) * DIM);
    dst[threadIdx.x] = src[threadIdx.x];   // 128 threads * 16B = 2048B = DIM bf16
}

// ---------------------------------------------------------------- WMMA GEMM
// Out[M,N] = A[M,K] * Bt[N,K]^T   (A row-major MxK bf16, Bt row-major NxK bf16)
// Block tile 128x256 (8 waves as 2x4); wave tile 64x64 -> 4x4 accum tiles.
// A/B tiles staged in LDS (cooperative b128 loads); fragments via ds_load_b128.
// MODE 0: Hout[row*ldh+col] = bf16(gelu(acc))
// MODE 1: tok=tok_slot[z*CAP+row]; if(tok>=0) atomicAdd(Y[tok*Dout+col], acc*gate)
// MODE 2: atomicAdd(Y[row*Dout+col], acc*gshared[row])
template <int MODE>
__global__ __launch_bounds__(256) void gemm_bf16_kernel(
    const __bf16* __restrict__ A, const __bf16* __restrict__ Bt, int Kt,
    size_t strideA, size_t strideB,
    __bf16* __restrict__ Hout, size_t strideH, int ldh,
    float* __restrict__ Y, const int* __restrict__ tok_slot,
    const float* __restrict__ gate_slot, int Dout,
    const float* __restrict__ gshared)
{
    __shared__ __bf16 As[128 * LDS_STRIDE];   // 10 KB
    __shared__ __bf16 Bs[256 * LDS_STRIDE];   // 20 KB

    const int z = blockIdx.z;
    A  += (size_t)z * strideA;
    Bt += (size_t)z * strideB;
    if (MODE == 0) Hout += (size_t)z * strideH;

    const int tid    = threadIdx.x;
    const int lane   = tid & 31;
    const int wave   = tid >> 5;
    const int waveM  = wave >> 2;          // 0..1
    const int waveN  = wave & 3;           // 0..3
    const int r      = lane & 15;
    const int h      = lane >> 4;          // K-half for A/B fragments
    const int blockRow = blockIdx.x * 128;
    const int blockCol = blockIdx.y * 256;

    v8f acc[4][4];
    #pragma unroll
    for (int i = 0; i < 4; i++)
        #pragma unroll
        for (int j = 0; j < 4; j++) acc[i][j] = (v8f)(0.f);

    union Frag { v16bf v; v8bf half[2]; };

    for (int k = 0; k < Kt; k += 32) {
        __syncthreads();   // protect LDS reads of previous iteration
        // --- cooperative global -> LDS: A tile 128x32 (512 x 16B chunks) ---
        #pragma unroll
        for (int c = 0; c < 2; c++) {
            int q   = tid + c * 256;
            int row = q >> 2, kc = (q & 3) * 8;
            *(uint4*)(&As[row * LDS_STRIDE + kc]) =
                *(const uint4*)(A + (size_t)(blockRow + row) * Kt + k + kc);
        }
        // --- cooperative global -> LDS: B tile 256x32 (1024 x 16B chunks) ---
        #pragma unroll
        for (int c = 0; c < 4; c++) {
            int q   = tid + c * 256;
            int row = q >> 2, kc = (q & 3) * 8;
            *(uint4*)(&Bs[row * LDS_STRIDE + kc]) =
                *(const uint4*)(Bt + (size_t)(blockCol + row) * Kt + k + kc);
        }
        __syncthreads();

        // --- fragments from LDS ---
        Frag afr[4], bfr[4];
        #pragma unroll
        for (int i = 0; i < 4; i++) {
            const __bf16* p = &As[(waveM * 64 + i * 16 + r) * LDS_STRIDE + 16 * h];
            afr[i].half[0] = *(const v8bf*)(p);
            afr[i].half[1] = *(const v8bf*)(p + 8);
        }
        #pragma unroll
        for (int j = 0; j < 4; j++) {
            const __bf16* p = &Bs[(waveN * 64 + j * 16 + r) * LDS_STRIDE + 16 * h];
            bfr[j].half[0] = *(const v8bf*)(p);
            bfr[j].half[1] = *(const v8bf*)(p + 8);
        }
        #pragma unroll
        for (int i = 0; i < 4; i++)
            #pragma unroll
            for (int j = 0; j < 4; j++)
                acc[i][j] = __builtin_amdgcn_wmma_f32_16x16x32_bf16(
                    false, afr[i].v, false, bfr[j].v, (short)0, acc[i][j], false, false);
    }

    #pragma unroll
    for (int i = 0; i < 4; i++) {
        #pragma unroll
        for (int j = 0; j < 4; j++) {
            int col = blockCol + waveN * 64 + j * 16 + r;
            #pragma unroll
            for (int v = 0; v < 8; v++) {
                int row = blockRow + waveM * 64 + i * 16 + h * 8 + v;
                float val = acc[i][j][v];
                if constexpr (MODE == 0) {
                    float g = 0.5f * val * (1.0f + erff(val * 0.70710678118654752f));
                    Hout[(size_t)row * ldh + col] = (__bf16)g;
                } else if constexpr (MODE == 1) {
                    int slot = z * CAP + row;
                    int tok = tok_slot[slot];
                    if (tok >= 0)
                        atomicAdd(Y + (size_t)tok * Dout + col, val * gate_slot[slot]);
                } else {
                    atomicAdd(Y + (size_t)row * Dout + col, val * gshared[row]);
                }
            }
        }
    }
}

// ---------------------------------------------------------------- finalize
__global__ void finalize_kernel(const float* __restrict__ imp, const int* __restrict__ counts,
                                float* __restrict__ out_imp, float* __restrict__ out_load)
{
    int e = threadIdx.x;                 // blockDim = 32
    __shared__ float total;
    if (e == 0) {
        int t = 0;
        for (int i = 0; i < E_EXPERTS; i++) t += counts[i];
        total = (float)t;
    }
    __syncthreads();
    if (e < E_EXPERTS) {
        out_imp[e]  = imp[e] * (1.0f / (float)NTOK);
        out_load[e] = (float)counts[e] / (total + 1e-12f);
    }
}

// ---------------------------------------------------------------- launcher
extern "C" void kernel_launch(void* const* d_in, const int* in_sizes, int n_in,
                              void* d_out, int out_size, void* d_ws, size_t ws_size,
                              hipStream_t stream)
{
    const float* x    = (const float*)d_in[0];   // (4,2048,1024)
    const float* Wr   = (const float*)d_in[1];   // (1024,16)
    const float* Wg   = (const float*)d_in[2];   // (1024,1)
    const float* w1   = (const float*)d_in[3];   // (16,1024,512)
    const float* w2   = (const float*)d_in[4];   // (16,512,1024)
    const float* sw1  = (const float*)d_in[5];   // (1024,512)
    const float* sw2  = (const float*)d_in[6];   // (512,1024)

    float* y        = (float*)d_out;                       // N*D
    float* out_imp  = y + (size_t)NTOK * DIM;              // 16
    float* out_load = out_imp + E_EXPERTS;                 // 16

    // ---- workspace layout (256B aligned slices) ----
    char* ws = (char*)d_ws;
    size_t off = 0;
    auto carve = [&](size_t bytes) { size_t o = off; off = (off + bytes + 255) & ~(size_t)255; return o; };
    __bf16* W1T   = (__bf16*)(ws + carve((size_t)E_EXPERTS * FFE * DIM * 2));   // (E, F, D)
    __bf16* W2T   = (__bf16*)(ws + carve((size_t)E_EXPERTS * DIM * FFE * 2));   // (E, D, F)
    __bf16* SW1T  = (__bf16*)(ws + carve((size_t)DFFS * DIM * 2));              // (DFF, D)
    __bf16* SW2T  = (__bf16*)(ws + carve((size_t)DIM * DFFS * 2));              // (D, DFF)
    __bf16* XBF   = (__bf16*)(ws + carve((size_t)NTOK * DIM * 2));              // (N, D)
    __bf16* BUF   = (__bf16*)(ws + carve((size_t)E_EXPERTS * CAP * DIM * 2));   // (E, C, D)
    __bf16* H1    = (__bf16*)(ws + carve((size_t)E_EXPERTS * CAP * FFE * 2));   // (E, C, F)
    __bf16* HSH   = (__bf16*)(ws + carve((size_t)NTOK * DFFS * 2));             // (N, DFF)
    int*    ROUTE_E   = (int*)  (ws + carve((size_t)NTOK * TOPK * 4));
    int*    ROUTE_POS = (int*)  (ws + carve((size_t)NTOK * TOPK * 4));
    float*  ROUTE_G   = (float*)(ws + carve((size_t)NTOK * TOPK * 4));
    float*  GSHARED   = (float*)(ws + carve((size_t)NTOK * 4));
    float*  IMP       = (float*)(ws + carve(E_EXPERTS * 4));
    int*    COUNTS    = (int*)  (ws + carve(E_EXPERTS * 4));
    int*    TOK_SLOT  = (int*)  (ws + carve((size_t)E_EXPERTS * CAP * 4));
    float*  GATE_SLOT = (float*)(ws + carve((size_t)E_EXPERTS * CAP * 4));

    // ---- init ----
    fill_u32_kernel<<<4096, 256, 0, stream>>>((unsigned*)y, 0u, (size_t)NTOK * DIM);
    fill_u32_kernel<<<1, 64, 0, stream>>>((unsigned*)IMP, 0u, E_EXPERTS);
    fill_u32_kernel<<<1, 64, 0, stream>>>((unsigned*)COUNTS, 0u, E_EXPERTS);
    fill_u32_kernel<<<4096, 256, 0, stream>>>((unsigned*)BUF, 0u, (size_t)E_EXPERTS * CAP * DIM / 2);
    fill_u32_kernel<<<80, 256, 0, stream>>>((unsigned*)TOK_SLOT, 0xFFFFFFFFu, (size_t)E_EXPERTS * CAP);

    // ---- prep: bf16 conversions + weight transposes to N-major ----
    f32_to_bf16_kernel<<<4096, 256, 0, stream>>>(x, XBF, (size_t)NTOK * DIM);
    transpose_to_bf16_kernel<<<dim3(FFE / 32, DIM / 32, E_EXPERTS), dim3(32, 8), 0, stream>>>(
        w1, W1T, DIM, FFE, (size_t)DIM * FFE, (size_t)DIM * FFE);
    transpose_to_bf16_kernel<<<dim3(DIM / 32, FFE / 32, E_EXPERTS), dim3(32, 8), 0, stream>>>(
        w2, W2T, FFE, DIM, (size_t)FFE * DIM, (size_t)FFE * DIM);
    transpose_to_bf16_kernel<<<dim3(DFFS / 32, DIM / 32, 1), dim3(32, 8), 0, stream>>>(
        sw1, SW1T, DIM, DFFS, 0, 0);
    transpose_to_bf16_kernel<<<dim3(DIM / 32, DFFS / 32, 1), dim3(32, 8), 0, stream>>>(
        sw2, SW2T, DFFS, DIM, 0, 0);

    // ---- routing ----
    router_kernel<<<NTOK / 8, 256, 0, stream>>>(x, Wr, Wg, ROUTE_E, ROUTE_G, GSHARED, IMP, COUNTS);
    position_kernel<<<E_EXPERTS, 32, 0, stream>>>(ROUTE_E, ROUTE_G, ROUTE_POS, TOK_SLOT, GATE_SLOT);
    gather_kernel<<<NTOK * TOPK, 128, 0, stream>>>(XBF, ROUTE_E, ROUTE_POS, BUF);

    // ---- expert FFN: up-proj + GELU, then down-proj fused with gated scatter ----
    gemm_bf16_kernel<0><<<dim3(CAP / 128, FFE / 256, E_EXPERTS), 256, 0, stream>>>(
        BUF, W1T, DIM, (size_t)CAP * DIM, (size_t)FFE * DIM,
        H1, (size_t)CAP * FFE, FFE, nullptr, nullptr, nullptr, 0, nullptr);
    gemm_bf16_kernel<1><<<dim3(CAP / 128, DIM / 256, E_EXPERTS), 256, 0, stream>>>(
        H1, W2T, FFE, (size_t)CAP * FFE, (size_t)DIM * FFE,
        nullptr, 0, 0, y, TOK_SLOT, GATE_SLOT, DIM, nullptr);

    // ---- shared expert: up-proj + GELU, down-proj fused with g_shared scatter ----
    gemm_bf16_kernel<0><<<dim3(NTOK / 128, DFFS / 256, 1), 256, 0, stream>>>(
        XBF, SW1T, DIM, 0, 0,
        HSH, 0, DFFS, nullptr, nullptr, nullptr, 0, nullptr);
    gemm_bf16_kernel<2><<<dim3(NTOK / 128, DIM / 256, 1), 256, 0, stream>>>(
        HSH, SW2T, DFFS, 0, 0,
        nullptr, 0, 0, y, nullptr, nullptr, DIM, GSHARED);

    // ---- stats ----
    finalize_kernel<<<1, 32, 0, stream>>>(IMP, COUNTS, out_imp, out_load);
}
